// OrderEdgeConv_81475529605939
// MI455X (gfx1250) — matrix operand order repaired and checked by reading, not compiled, not run on hardware
//
#include <hip/hip_runtime.h>
#include <hip/hip_bf16.h>

typedef unsigned short ushort_t;
typedef __attribute__((ext_vector_type(16))) __bf16 v16bf;
typedef __attribute__((ext_vector_type(2)))  __bf16 v2bf;
typedef __attribute__((ext_vector_type(8)))  float  v8f;

union Frag {
    uint4 u4[2];
    v16bf v;
};

// ---- scalar / packed f32 <-> bf16 helpers -------------------------------
__device__ __forceinline__ ushort_t f2bf_sw(float f) {
    union { float f; unsigned u; } v; v.f = f;
    unsigned r = v.u + 0x7FFFu + ((v.u >> 16) & 1u);   // RNE
    return (ushort_t)(r >> 16);
}
__device__ __forceinline__ unsigned pk_bf16(float a, float b) {
#if __has_builtin(__builtin_amdgcn_cvt_pk_bf16_f32)
    v2bf r = __builtin_amdgcn_cvt_pk_bf16_f32(a, b);
    union { v2bf v; unsigned u; } c; c.v = r; return c.u;
#else
    return (unsigned)f2bf_sw(a) | ((unsigned)f2bf_sw(b) << 16);
#endif
}
__device__ __forceinline__ ushort_t f2bf(float v) {
    return (ushort_t)(pk_bf16(v, v) & 0xFFFFu);
}
__device__ __forceinline__ float bf2f(ushort_t h) {
    union { unsigned u; float f; } v; v.u = ((unsigned)h) << 16;
    return v.f;
}
// packed bf16 max (V_PK_MAX_NUM_BF16, VOP3P op 44)
__device__ __forceinline__ unsigned pk_max_bf16(unsigned a, unsigned b) {
    unsigned d;
    asm volatile("v_pk_max_num_bf16 %0, %1, %2" : "=v"(d) : "v"(a), "v"(b));
    return d;
}
__device__ __forceinline__ void wave_lds_fence() {
    __builtin_amdgcn_wave_barrier();
    asm volatile("" ::: "memory");
}

// ---------------------------------------------------------------------------
// Prep: W1..W4 (f32 [Kin][64]) -> bf16 WMMA B-fragment layout in ws.
// B 32x16 bf16 (wave32): lane L holds col n=L%16(+16t), b[i]=W[k0+16*(L/16)+i][n].
// Groups per layer: 24,16,24,32 (each group = 512 bf16 = one 32-lane fragment).
// ---------------------------------------------------------------------------
__global__ __launch_bounds__(256) void oec_prep_weights(
    const float* __restrict__ W1, const float* __restrict__ W2,
    const float* __restrict__ W3, const float* __restrict__ W4,
    ushort_t* __restrict__ ws)
{
    int e = blockIdx.x * 256 + threadIdx.x;     // 0 .. 49151
    int g   = e >> 9;
    int rem = e & 511;
    int L   = rem >> 4;
    int i   = rem & 15;

    const float* W; int g0;
    if      (g < 24) { W = W1; g0 = 0;  }
    else if (g < 40) { W = W2; g0 = 24; }
    else if (g < 64) { W = W3; g0 = 40; }
    else             { W = W4; g0 = 64; }
    int gl = g - g0;
    int s  = gl >> 2;
    int t  = gl & 3;
    int k  = 32 * s + 16 * (L >> 4) + i;
    int n  = 16 * t + (L & 15);
    ws[e] = f2bf_sw(W[k * 64 + n]);
}

// ---------------------------------------------------------------------------
// Panel per point: 16 rows x 384 bf16 cols:
//   [0:64)=h3 [64:128)=h2 [128:192)=h1 [192:256)=xt [256:320)=knn [320:384)=ord
// Layer input windows: ef=[192:384), a2=[128:256), a3=[64:256), a4=[0:256).
// Final feature = [h4(regs) | cols 0:256) ] -> out channels [0:64) | [64:320).
// ---------------------------------------------------------------------------
constexpr int PCOLS = 384;
constexpr int PSZ   = 16 * PCOLS;        // ushorts per point panel

// One dense layer over 4 points sharing B fragments in registers.
// Per k-step: 8 global_load_b128 (B, shared) + 8 ds_load_b128 (A, batched)
// feeding 16 v_wmma_f32_16x16x32_bf16 (16 independent accumulation chains).
template<int KIN, int ACOL, int WSOFF, bool STORE_RELU>
__device__ __forceinline__ void run_layer4p(
    ushort_t* __restrict__ lds0,              // wave's 4-panel base
    const ushort_t* __restrict__ wsb,
    const float* __restrict__ bias,
    int lane, v8f acc[4][4])                  // [point][tile]
{
    const int m  = lane & 15;
    const int hl = lane >> 4;

#pragma unroll
    for (int p = 0; p < 4; ++p)
#pragma unroll
        for (int t = 0; t < 4; ++t)
#pragma unroll
            for (int r = 0; r < 8; ++r) acc[p][t][r] = 0.0f;

#pragma unroll
    for (int s = 0; s < KIN / 32; ++s) {
        Frag B[4];
#pragma unroll
        for (int t = 0; t < 4; ++t) {
            const uint4* bq =
                (const uint4*)(wsb + WSOFF + (((s * 4 + t) * 32 + lane) << 4));
            B[t].u4[0] = bq[0];
            B[t].u4[1] = bq[1];
        }
        Frag A[4];
#pragma unroll
        for (int p = 0; p < 4; ++p) {
            const ushort_t* arow = lds0 + p * PSZ + m * PCOLS + ACOL + 32 * s + 16 * hl;
            A[p].u4[0] = *(const uint4*)(arow);
            A[p].u4[1] = *(const uint4*)(arow + 8);
        }
#pragma unroll
        for (int p = 0; p < 4; ++p) {
#pragma unroll
            for (int t = 0; t < 4; ++t) {
                acc[p][t] = __builtin_amdgcn_wmma_f32_16x16x32_bf16(
                    false, A[p].v, false, B[t].v, (short)0, acc[p][t], false, false);
            }
        }
    }

#pragma unroll
    for (int t = 0; t < 4; ++t) {
        float bv = bias[16 * t + m];
#pragma unroll
        for (int p = 0; p < 4; ++p) {
#pragma unroll
            for (int r = 0; r < 8; ++r) {
                float v = acc[p][t][r] + bv;
                if (STORE_RELU) {
                    v = fmaxf(v, 0.0f);
                    lds0[p * PSZ + (8 * hl + r) * PCOLS + (ACOL - 64) + 16 * t + m] = f2bf(v);
                } else {
                    acc[p][t][r] = v;
                }
            }
        }
    }
    if (STORE_RELU) wave_lds_fence();
}

// ---------------------------------------------------------------------------
// Main: 64 threads = 2 waves; each wave batches 4 points per layer pass so
// every B fragment load feeds 4 WMMAs (4x B-traffic reduction vs 1 pt/wave).
// Dynamic LDS: 2 waves * 4 points * 12KB = 96KB.
// ---------------------------------------------------------------------------
__global__ __launch_bounds__(64) void oec_main(
    const float* __restrict__ x,
    const float* __restrict__ b1, const float* __restrict__ b2,
    const float* __restrict__ b3, const float* __restrict__ b4,
    const ushort_t* __restrict__ wsb,
    float* __restrict__ out)
{
    constexpr int N = 8192;
    extern __shared__ ushort_t LDSB[];

    const int wave = threadIdx.x >> 5;
    const int lane = threadIdx.x & 31;
    ushort_t* lds0 = LDSB + wave * 4 * PSZ;

#pragma unroll 1
    for (int it = 0; it < 4; ++it) {
        const int pbase = blockIdx.x * 32 + wave * 16 + it * 4;   // 4 consecutive pts

        // ---- prefetch next group's x rows toward the WGP (fire-and-forget)
        if (it < 3) {
            const int pn = pbase + 4;
            const int bn = pn >> 13;
            const int nn = pn & (N - 1);
            int r0 = nn - 8; r0 = r0 < 0 ? 0 : r0;
            const char* pf = (const char*)(x + ((size_t)bn * N + r0) * 64);
            __builtin_prefetch(pf + lane * 160, 0, 3);   // ~5KB span: rows n-8..n+11
        }

        // ---- build edge features for 4 points: lanes (2r,2r+1) fill row r
#pragma unroll 1
        for (int p = 0; p < 4; ++p) {
            const int pp = pbase + p;
            const int b  = pp >> 13;
            const int n  = pp & (N - 1);
            const float* __restrict__ xb = x + ((size_t)b * N) * 64;

            const int row   = lane >> 1;
            const int cbase = (lane & 1) * 32;

            int off = (row < 8) ? (row - 8) : (row - 7);
            int nb  = n - off;
            nb = nb < 0 ? 0 : (nb > N - 1 ? N - 1 : nb);

            const float4* c4 = (const float4*)(xb + (size_t)n  * 64 + cbase);
            const float4* n4 = (const float4*)(xb + (size_t)nb * 64 + cbase);

            float cf[32], nf[32];
#pragma unroll
            for (int q = 0; q < 8; ++q) {
                float4 cv = c4[q], nv = n4[q];
                cf[4 * q + 0] = cv.x; cf[4 * q + 1] = cv.y;
                cf[4 * q + 2] = cv.z; cf[4 * q + 3] = cv.w;
                nf[4 * q + 0] = nv.x; nf[4 * q + 1] = nv.y;
                nf[4 * q + 2] = nv.z; nf[4 * q + 3] = nv.w;
            }

            ushort_t* prow = lds0 + p * PSZ + row * PCOLS;
            unsigned* xt = (unsigned*)(prow + 192 + cbase);
            unsigned* kn = (unsigned*)(prow + 256 + cbase);
            unsigned* od = (unsigned*)(prow + 320 + cbase);
            const float sgn = (row < 8) ? 1.0f : -1.0f;    // xc-knn : knn-xc
#pragma unroll
            for (int i = 0; i < 16; ++i) {
                float c0 = cf[2 * i], c1 = cf[2 * i + 1];
                float k0 = nf[2 * i], k1 = nf[2 * i + 1];
                xt[i] = pk_bf16(c0, c1);
                kn[i] = pk_bf16(k0, k1);
                od[i] = pk_bf16(sgn * (c0 - k0), sgn * (c1 - k1));
            }
        }
        wave_lds_fence();

        // ---- 4 dense layers, B fragments shared across the 4 points
        v8f acc[4][4];
        run_layer4p<192, 192, 0,        true >(lds0, wsb, b1, lane, acc);
        run_layer4p<128, 128, 24 * 512, true >(lds0, wsb, b2, lane, acc);
        run_layer4p<192,  64, 40 * 512, true >(lds0, wsb, b3, lane, acc);
        run_layer4p<256,   0, 64 * 512, false>(lds0, wsb, b4, lane, acc);

        // ---- per-point max over 16 rows + store
#pragma unroll 1
        for (int p = 0; p < 4; ++p) {
            float* __restrict__ outp = out + (size_t)(pbase + p) * 320;

            // h4 (registers): lane holds col n=L%16+16t, rows m=8*(L/16)+r
#pragma unroll
            for (int t = 0; t < 4; ++t) {
                float mx = acc[p][t][0];
#pragma unroll
                for (int r = 1; r < 8; ++r) mx = fmaxf(mx, acc[p][t][r]);
                mx = fmaxf(mx, __shfl_xor(mx, 16, 32));
                if (lane < 16) outp[16 * t + lane] = mx;
            }

            // panel cols [0:256): packed bf16 max across 16 rows, 8 cols/lane
            const ushort_t* pp0 = lds0 + p * PSZ + lane * 8;
            unsigned mx[4] = {0xFF7FFF7Fu, 0xFF7FFF7Fu, 0xFF7FFF7Fu, 0xFF7FFF7Fu};
#pragma unroll
            for (int row = 0; row < 16; ++row) {
                uint4 v = *(const uint4*)(pp0 + row * PCOLS);
                mx[0] = pk_max_bf16(mx[0], v.x);
                mx[1] = pk_max_bf16(mx[1], v.y);
                mx[2] = pk_max_bf16(mx[2], v.z);
                mx[3] = pk_max_bf16(mx[3], v.w);
            }
            float4 o0 = make_float4(bf2f((ushort_t)(mx[0] & 0xFFFF)), bf2f((ushort_t)(mx[0] >> 16)),
                                    bf2f((ushort_t)(mx[1] & 0xFFFF)), bf2f((ushort_t)(mx[1] >> 16)));
            float4 o1 = make_float4(bf2f((ushort_t)(mx[2] & 0xFFFF)), bf2f((ushort_t)(mx[2] >> 16)),
                                    bf2f((ushort_t)(mx[3] & 0xFFFF)), bf2f((ushort_t)(mx[3] >> 16)));
            *(float4*)&outp[64 + lane * 8]     = o0;
            *(float4*)&outp[64 + lane * 8 + 4] = o1;
        }
        wave_lds_fence();   // panels reused next iteration
    }
}

extern "C" void kernel_launch(void* const* d_in, const int* in_sizes, int n_in,
                              void* d_out, int out_size, void* d_ws, size_t ws_size,
                              hipStream_t stream) {
    const float* x   = (const float*)d_in[0];
    // d_in[1] = pos (unused by the reference computation)
    const float* W1  = (const float*)d_in[2];
    const float* b1  = (const float*)d_in[3];
    const float* W2  = (const float*)d_in[4];
    const float* b2  = (const float*)d_in[5];
    const float* W3  = (const float*)d_in[6];
    const float* b3  = (const float*)d_in[7];
    const float* W4  = (const float*)d_in[8];
    const float* b4  = (const float*)d_in[9];
    float* out       = (float*)d_out;
    ushort_t* ws     = (ushort_t*)d_ws;

    oec_prep_weights<<<192, 256, 0, stream>>>(W1, W2, W3, W4, ws);

    const int smem = 2 * 4 * PSZ * (int)sizeof(ushort_t);   // 96 KB dynamic LDS
    (void)hipFuncSetAttribute((const void*)oec_main,
                              hipFuncAttributeMaxDynamicSharedMemorySize, smem);
    // 32768 points, 32 per block (2 waves x 4 points x 4 iterations)
    oec_main<<<1024, 64, smem, stream>>>(x, b1, b2, b3, b4, ws, out);
}
